// OutlookAttention_60224031425234
// MI455X (gfx1250) — matrix operand order repaired and checked by reading, not compiled, not run on hardware
//
#include <hip/hip_runtime.h>
#include <hip/hip_bf16.h>

// Problem constants (reference: B,H,W,C = 8,56,56,384; HEADS=12, K=3, PAD=1)
#define BATCH   8
#define HH      56
#define WW      56
#define CC      384
#define HEADS   12
#define DHEAD   32
#define K2      9
#define NPIX    (HH * WW)          // 3136
#define MROWS   (BATCH * NPIX)     // 25088  (multiple of 64)
#define NA      972                // K^4 * HEADS
#define NA_PAD  1024               // padded to multiple of 64 for GEMM tiling
#define LDA_A   976                // row stride of the logits buffer (>= NA, mult of 4)

typedef __attribute__((ext_vector_type(16))) __bf16 v16bf;
typedef __attribute__((ext_vector_type(8)))  float  v8f;

union FragAB { v16bf v; uint4 u[2]; };

// ---------------------------------------------------------------- casts ----
__global__ void cast_f32_bf16(const float* __restrict__ in,
                              __hip_bfloat16* __restrict__ out, int n) {
    int i = blockIdx.x * blockDim.x + threadIdx.x;
    if (i < n) out[i] = __float2bfloat16(in[i]);
}

__global__ void cast_pad_wa(const float* __restrict__ in,
                            __hip_bfloat16* __restrict__ out) {
    int i = blockIdx.x * blockDim.x + threadIdx.x;
    const int total = NA_PAD * CC;
    if (i >= total) return;
    int row = i / CC;
    out[i] = (row < NA) ? __float2bfloat16(in[i]) : __float2bfloat16(0.0f);
}

// ------------------------------------------------------ WMMA bf16 GEMM ----
// Out[m,n] = sum_k X[m,k] * W[n,k]   (X: M x K row-major, W: N x K row-major)
// Block = 128 threads (4 waves), block tile 64x64, wave tile 32x32
// (2x2 grid of v_wmma_f32_16x16x32_bf16 tiles, K stepped by 32).

__device__ inline void store_f32(float* Out, int ldo, int Nstore, int mt, int nt,
                                 int lhi, int llo, v8f acc) {
    int col = nt + llo;
    if (col >= Nstore) return;
#pragma unroll
    for (int r = 0; r < 8; ++r)
        Out[(size_t)(mt + lhi * 8 + r) * ldo + col] = acc[r];
}

__device__ inline void store_bf16(__hip_bfloat16* Out, int ldo, int Nstore, int mt,
                                  int nt, int lhi, int llo, v8f acc) {
    int col = nt + llo;
    if (col >= Nstore) return;
#pragma unroll
    for (int r = 0; r < 8; ++r)
        Out[(size_t)(mt + lhi * 8 + r) * ldo + col] = __float2bfloat16(acc[r]);
}

__global__ __launch_bounds__(128)
void wmma_gemm_bf16(const __hip_bfloat16* __restrict__ X,
                    const __hip_bfloat16* __restrict__ W,
                    void* __restrict__ OutP,
                    int K, int Nstore, int ldo, int out_is_bf16) {
    const int lane = threadIdx.x & 31;
    const int wave = threadIdx.x >> 5;
    const int lhi  = lane >> 4;   // 0/1: which half-wave
    const int llo  = lane & 15;

    const int m0 = blockIdx.x * 64 + (wave >> 1) * 32;
    const int n0 = blockIdx.y * 64 + (wave & 1) * 32;

    // A: lane holds row M = m0(+16) + llo; chunk0 @K=lhi*8, chunk1 @K=16+lhi*8
    const __hip_bfloat16* a0p = X + (size_t)(m0 + llo)      * K;
    const __hip_bfloat16* a1p = X + (size_t)(m0 + 16 + llo) * K;
    // B: lane holds column N = n0(+16) + llo = row of W; chunk0 @K=lhi*8, chunk1 @K=16+lhi*8
    const __hip_bfloat16* b0p = W + (size_t)(n0 + llo)      * K;
    const __hip_bfloat16* b1p = W + (size_t)(n0 + 16 + llo) * K;

    v8f acc00 = {}, acc01 = {}, acc10 = {}, acc11 = {};

    for (int k0 = 0; k0 < K; k0 += 32) {
        FragAB a0, a1, b0, b1;
        const int ka = k0 + lhi * 8;
        a0.u[0] = *(const uint4*)(a0p + ka);
        a0.u[1] = *(const uint4*)(a0p + ka + 16);
        a1.u[0] = *(const uint4*)(a1p + ka);
        a1.u[1] = *(const uint4*)(a1p + ka + 16);
        const int kb = k0 + lhi * 8;
        b0.u[0] = *(const uint4*)(b0p + kb);
        b0.u[1] = *(const uint4*)(b0p + kb + 16);
        b1.u[0] = *(const uint4*)(b1p + kb);
        b1.u[1] = *(const uint4*)(b1p + kb + 16);

        acc00 = __builtin_amdgcn_wmma_f32_16x16x32_bf16(false, a0.v, false, b0.v,
                                                        (short)0, acc00, false, false);
        acc01 = __builtin_amdgcn_wmma_f32_16x16x32_bf16(false, a0.v, false, b1.v,
                                                        (short)0, acc01, false, false);
        acc10 = __builtin_amdgcn_wmma_f32_16x16x32_bf16(false, a1.v, false, b0.v,
                                                        (short)0, acc10, false, false);
        acc11 = __builtin_amdgcn_wmma_f32_16x16x32_bf16(false, a1.v, false, b1.v,
                                                        (short)0, acc11, false, false);
    }

    if (out_is_bf16) {
        __hip_bfloat16* Out = (__hip_bfloat16*)OutP;
        store_bf16(Out, ldo, Nstore, m0,      n0,      lhi, llo, acc00);
        store_bf16(Out, ldo, Nstore, m0,      n0 + 16, lhi, llo, acc01);
        store_bf16(Out, ldo, Nstore, m0 + 16, n0,      lhi, llo, acc10);
        store_bf16(Out, ldo, Nstore, m0 + 16, n0 + 16, lhi, llo, acc11);
    } else {
        float* Out = (float*)OutP;
        store_f32(Out, ldo, Nstore, m0,      n0,      lhi, llo, acc00);
        store_f32(Out, ldo, Nstore, m0,      n0 + 16, lhi, llo, acc01);
        store_f32(Out, ldo, Nstore, m0 + 16, n0,      lhi, llo, acc10);
        store_f32(Out, ldo, Nstore, m0 + 16, n0 + 16, lhi, llo, acc11);
    }
}

// ------------------------------------------------------------- softmax ----
// softmax over q (9 contiguous floats) of a[row, head*81 + p*9 + q], scaled
// by d^-0.5 = 32^-0.5. One thread per (row, head, p).
__global__ void softmax9(float* __restrict__ A) {
    const int total = MROWS * HEADS * K2;
    int idx = blockIdx.x * blockDim.x + threadIdx.x;
    if (idx >= total) return;
    int row = idx / (HEADS * K2);
    int hp  = idx % (HEADS * K2);
    float* p = A + (size_t)row * LDA_A + (hp / K2) * (K2 * K2) + (hp % K2) * K2;
    const float scale = 0.17677669529663687f;   // 1/sqrt(32)
    float v[9], m = -3.4e38f;
#pragma unroll
    for (int q = 0; q < 9; ++q) { v[q] = p[q] * scale; m = fmaxf(m, v[q]); }
    float sum = 0.0f;
#pragma unroll
    for (int q = 0; q < 9; ++q) { v[q] = __expf(v[q] - m); sum += v[q]; }
    float inv = 1.0f / sum;
#pragma unroll
    for (int q = 0; q < 9; ++q) p[q] = v[q] * inv;
}

// ----------------------------------------- attention apply + fold (gather) -
// y[b,r,s,c] = sum_{i,j in 0..2, n'=(r+1-i, s+1-j) valid}
//                sum_q attn[b,n',head, p=i*3+j, q=(qi,qj)] * v[b, n'+q-1, c]
// One block per output pixel; 384 threads = 12 waves, wave <-> head, so the
// attn scalar load is wave-uniform and the V load is a 64B coalesced row.
__global__ __launch_bounds__(384)
void outlook_av_fold(const float* __restrict__ attn,
                     const __hip_bfloat16* __restrict__ V,
                     __hip_bfloat16* __restrict__ Y) {
    const int pix = blockIdx.x;              // b*NPIX + r*WW + s
    const int b   = pix / NPIX;
    const int rs  = pix % NPIX;
    const int r   = rs / WW, s = rs % WW;
    const int t    = threadIdx.x;            // channel c
    const int head = t >> 5;

    float acc = 0.0f;
    for (int i = 0; i < 3; ++i) {
        int ny = r + 1 - i;
        if ((unsigned)ny >= (unsigned)HH) continue;
        for (int j = 0; j < 3; ++j) {
            int nx = s + 1 - j;
            if ((unsigned)nx >= (unsigned)WW) continue;
            const int p = i * 3 + j;
            const float* arow = attn + (size_t)(b * NPIX + ny * WW + nx) * LDA_A
                                     + head * (K2 * K2) + p * K2;
            for (int qi = 0; qi < 3; ++qi) {
                int vy = ny + qi - 1;
                if ((unsigned)vy >= (unsigned)HH) continue;
                for (int qj = 0; qj < 3; ++qj) {
                    int vx = nx + qj - 1;
                    if ((unsigned)vx >= (unsigned)WW) continue;
                    float w   = arow[qi * 3 + qj];
                    float val = __bfloat162float(
                        V[(size_t)(b * NPIX + vy * WW + vx) * CC + t]);
                    acc += w * val;
                }
            }
        }
    }
    Y[(size_t)pix * CC + t] = __float2bfloat16(acc);
}

// -------------------------------------------------------------- launch ----
extern "C" void kernel_launch(void* const* d_in, const int* in_sizes, int n_in,
                              void* d_out, int out_size, void* d_ws, size_t ws_size,
                              hipStream_t stream) {
    const float* x  = (const float*)d_in[0];   // (8,56,56,384)
    const float* Wv = (const float*)d_in[1];   // (384,384)
    const float* Wa = (const float*)d_in[2];   // (972,384)
    const float* Wo = (const float*)d_in[3];   // (384,384)
    float* out = (float*)d_out;                // (8,56,56,384) fp32

    // Workspace carve-up (256B aligned).  Total ~138 MB.
    char*  ws  = (char*)d_ws;
    size_t off = 0;
    auto carve = [&](size_t bytes) {
        void* p = ws + off;
        off = (off + bytes + 255) & ~(size_t)255;
        return p;
    };
    __hip_bfloat16* Xbf  = (__hip_bfloat16*)carve((size_t)MROWS * CC * 2);
    __hip_bfloat16* Wvbf = (__hip_bfloat16*)carve((size_t)CC * CC * 2);
    __hip_bfloat16* Wobf = (__hip_bfloat16*)carve((size_t)CC * CC * 2);
    __hip_bfloat16* Wabf = (__hip_bfloat16*)carve((size_t)NA_PAD * CC * 2);
    __hip_bfloat16* Vbf  = (__hip_bfloat16*)carve((size_t)MROWS * CC * 2);
    float*          Alog = (float*)carve((size_t)MROWS * LDA_A * 4);
    __hip_bfloat16* Ybf  = Xbf;   // X is dead after GEMM1+GEMM2 (stream-ordered)

    // 1) casts
    {
        int n = MROWS * CC;
        cast_f32_bf16<<<(n + 255) / 256, 256, 0, stream>>>(x, Xbf, n);
        n = CC * CC;
        cast_f32_bf16<<<(n + 255) / 256, 256, 0, stream>>>(Wv, Wvbf, n);
        cast_f32_bf16<<<(n + 255) / 256, 256, 0, stream>>>(Wo, Wobf, n);
        n = NA_PAD * CC;
        cast_pad_wa<<<(n + 255) / 256, 256, 0, stream>>>(Wa, Wabf);
    }

    // 2) V = X * Wv^T  (bf16 out, 25088 x 384)
    wmma_gemm_bf16<<<dim3(MROWS / 64, CC / 64), 128, 0, stream>>>(
        Xbf, Wvbf, (void*)Vbf, CC, CC, CC, /*bf16_out=*/1);

    // 3) logits = X * Wa^T  (fp32, 25088 x 972, stride 976; N padded to 1024)
    wmma_gemm_bf16<<<dim3(MROWS / 64, NA_PAD / 64), 128, 0, stream>>>(
        Xbf, Wabf, (void*)Alog, CC, LDA_A, LDA_A, /*bf16_out=*/0);

    // 4) softmax over q
    {
        int n = MROWS * HEADS * K2;
        softmax9<<<(n + 255) / 256, 256, 0, stream>>>(Alog);
    }

    // 5) attention apply + 3x3 fold -> Y (bf16)
    outlook_av_fold<<<MROWS, 384, 0, stream>>>(Alog, Vbf, Ybf);

    // 6) out = Y * Wo^T  (fp32 -> d_out)
    wmma_gemm_bf16<<<dim3(MROWS / 64, CC / 64), 128, 0, stream>>>(
        Ybf, Wobf, (void*)out, CC, CC, CC, /*bf16_out=*/0);
}
